// TriangleAttentionStartingNode_58179626991876
// MI455X (gfx1250) — matrix operand order repaired and checked by reading, not compiled, not run on hardware
//
#include <hip/hip_runtime.h>

// ---------------------------------------------------------------------------
// TriangleAttentionStartingNode  (B=1, N=256, CZ=128, H=4, CH=32)
// bf16 WMMA pipeline, vectorized b128 data movement, double-buffered
// fragment fetch with sched barriers, hardware-rcp sigmoid/softmax.
// ---------------------------------------------------------------------------

typedef __bf16 bf16;
typedef __attribute__((ext_vector_type(16))) __bf16 v16bf;
typedef __attribute__((ext_vector_type(8)))  __bf16 v8bf;
typedef __attribute__((ext_vector_type(8)))  float  v8f;
typedef __attribute__((ext_vector_type(4)))  float  v4f;

#define CZ   128
#define NSEQ 256
#define NH   4
#define CH   32
#define NROW 65536           // NSEQ*NSEQ pair positions

// ---- WMMA wrapper ----------------------------------------------------------
__device__ __forceinline__ v8f wmma_bf16(v16bf a, v16bf b, v8f c) {
  return __builtin_amdgcn_wmma_f32_16x16x32_bf16(false, a, false, b,
                                                 (short)0, c, false, false);
}

__device__ __forceinline__ v16bf join16(v8bf lo, v8bf hi) {
  union { v16bf v; v8bf h[2]; } u;
  u.h[0] = lo; u.h[1] = hi;
  return u.v;
}

// A-matrix 16x32 bf16 fragment from row-major source, ld in elements.
__device__ __forceinline__ v16bf load_a_frag(const bf16* base, int ld) {
  int lane = threadIdx.x & 31;
  const bf16* r = base + (size_t)(lane & 15) * ld + (lane >> 4) * 8;
  return join16(*(const v8bf*)r, *(const v8bf*)(r + 16));
}

// B = M^T where M is row-major [N][K] (16-col tile, 32-K tile), ld in elems.
__device__ __forceinline__ v16bf load_bT_frag(const bf16* base, int ld) {
  int lane = threadIdx.x & 31;
  const bf16* r = base + (size_t)(lane & 15) * ld + (lane >> 4) * 8;
  return join16(*(const v8bf*)r, *(const v8bf*)(r + 16));
}

__device__ __forceinline__ float fast_rcp(float x) {
  return __builtin_amdgcn_rcpf(x);       // single v_rcp_f32
}
__device__ __forceinline__ float fast_sigmoid(float x) {
  return fast_rcp(1.f + __expf(-x));
}

// ---------------------------------------------------------------------------
// Kernel 0: weight prep (f32 -> bf16, transposed)
// ---------------------------------------------------------------------------
__global__ __launch_bounds__(256) void k_prep_w(
    const float* __restrict__ wq, const float* __restrict__ wk,
    const float* __restrict__ wv, const float* __restrict__ wg,
    const float* __restrict__ wo, bf16* __restrict__ wcatT,
    bf16* __restrict__ woT) {
  int idx = blockIdx.x * 256 + threadIdx.x;
  if (idx < 512 * CZ) {
    int col = idx >> 7, c = idx & 127;            // col: output, c: in-ch
    int sel = col >> 7, j = col & 127;            // j = h*32+d
    const float* src = (sel == 0) ? wq : (sel == 1) ? wk : (sel == 2) ? wv : wg;
    wcatT[idx] = (bf16)src[c * CZ + j];
  } else if (idx < 512 * CZ + CZ * CZ) {
    int k = idx - 512 * CZ;
    int out = k >> 7, cp = k & 127;               // cp = h*32+d
    int h = cp >> 5, d = cp & 31;
    woT[k] = (bf16)wo[(d * NH + h) * CZ + out];   // permute row order
  }
}

// ---------------------------------------------------------------------------
// Kernel 1: LayerNorm + QKVG projection + pair bias
//   proj_qk [65536][256] bf16  (cols 0-127 q, 128-255 k, in h*32+d order)
//   vt      [128][65536] bf16  (column-major V)
//   gt      [128][65536] bf16  (column-major sigmoid(gate))
//   bt      [4][256][256] f32  (bias transposed: [h][key][query])
// ---------------------------------------------------------------------------
__global__ __launch_bounds__(256) void k_ln_proj(
    const float* __restrict__ pair, const float* __restrict__ lnw,
    const float* __restrict__ lnb, const float* __restrict__ wb,
    const bf16* __restrict__ wcatT, float* __restrict__ bt,
    bf16* __restrict__ proj_qk, bf16* __restrict__ vt,
    bf16* __restrict__ gt) {
  extern __shared__ char smem[];
  bf16* xs = (bf16*)smem;                 // [128][128] bf16 = 32 KB
  int m0 = blockIdx.x * 128;
  int tid = threadIdx.x;

  // ---- LayerNorm: one thread per row, float4 streams ----
  if (tid < 128) {
    const v4f* row4 = (const v4f*)(pair + (size_t)(m0 + tid) * CZ);
    float s = 0.f, s2 = 0.f;
#pragma unroll
    for (int c4 = 0; c4 < 32; ++c4) {
      v4f v = row4[c4];
#pragma unroll
      for (int e = 0; e < 4; ++e) { s += v[e]; s2 += v[e] * v[e]; }
    }
    float mu  = s * (1.0f / CZ);
    float var = s2 * (1.0f / CZ) - mu * mu;
    float inv = rsqrtf(var + 1e-5f);
#pragma unroll
    for (int c8 = 0; c8 < 16; ++c8) {
      v4f a = row4[c8 * 2], b = row4[c8 * 2 + 1];
      v4f w0 = *(const v4f*)(lnw + c8 * 8),  w1 = *(const v4f*)(lnw + c8 * 8 + 4);
      v4f o0 = *(const v4f*)(lnb + c8 * 8),  o1 = *(const v4f*)(lnb + c8 * 8 + 4);
      v8bf pk;
#pragma unroll
      for (int e = 0; e < 4; ++e) pk[e]     = (bf16)((a[e] - mu) * inv * w0[e] + o0[e]);
#pragma unroll
      for (int e = 0; e < 4; ++e) pk[4 + e] = (bf16)((b[e] - mu) * inv * w1[e] + o1[e]);
      *(v8bf*)(xs + tid * CZ + c8 * 8) = pk;
    }
  }
  __syncthreads();

  // ---- pair bias: x @ wb -> bt[h][key][query] ----
  if (tid < 128) {
    v4f b4 = {0.f, 0.f, 0.f, 0.f};
    for (int c = 0; c < CZ; c += 8) {
      v8bf xv = *(const v8bf*)(xs + tid * CZ + c);
#pragma unroll
      for (int e = 0; e < 8; ++e) {
        v4f w = *(const v4f*)(wb + (c + e) * 4);
        b4 += w * (float)xv[e];
      }
    }
    int mg = m0 + tid;
    int jq = mg >> 8, kk = mg & 255;
#pragma unroll
    for (int h = 0; h < NH; ++h) bt[h * NROW + kk * 256 + jq] = b4[h];
  }

  // ---- GEMM: [128 x 128] @ [128 x 512], wave-per-16-row tile ----
  int wave = tid >> 5, lane = tid & 31;
  int n = lane & 15, half = lane >> 4;
  bf16* stage = xs + wave * 16 * CZ;      // wave-private 512B (reuses xs rows)
  v16bf afr[4];
#pragma unroll
  for (int kc = 0; kc < 4; ++kc)
    afr[kc] = load_a_frag(xs + wave * 16 * CZ + kc * 32, CZ);
  __syncthreads();   // bias readers done before staging overwrites xs

#pragma unroll 2
  for (int nt = 0; nt < 32; ++nt) {
    // one clause of 8 b128 loads; barrier keeps fragments in distinct regs
    v16bf bfr[4];
#pragma unroll
    for (int kc = 0; kc < 4; ++kc)
      bfr[kc] = load_bT_frag(wcatT + (size_t)(nt * 16) * CZ + kc * 32, CZ);
    __builtin_amdgcn_sched_barrier(0);
    v8f acc = {};
#pragma unroll
    for (int kc = 0; kc < 4; ++kc) acc = wmma_bf16(afr[kc], bfr[kc], acc);

    int mg0 = m0 + wave * 16 + 8 * half;  // 8 consecutive rows per lane
    if (nt < 16) {
      // q/k: row-major store via wave-private LDS transpose
#pragma unroll
      for (int r = 0; r < 8; ++r)
        stage[(r + 8 * half) * 16 + n] = (bf16)acc[r];
      int m = lane & 15, h2 = lane >> 4;
      v8bf row = *(const v8bf*)(stage + m * 16 + h2 * 8);
      *(v8bf*)(proj_qk + (size_t)(m0 + wave * 16 + m) * 256 + nt * 16 + h2 * 8)
          = row;
    } else if (nt < 24) {
      // v: column-major packed store (8 consecutive rows / lane)
      int c2 = (nt - 16) * 16 + n;
      v8bf pk;
#pragma unroll
      for (int r = 0; r < 8; ++r) pk[r] = (bf16)acc[r];
      *(v8bf*)(vt + (size_t)c2 * NROW + mg0) = pk;
    } else {
      // gate: sigmoid (v_exp + v_rcp), column-major packed store
      int c2 = (nt - 24) * 16 + n;
      v8bf pk;
#pragma unroll
      for (int r = 0; r < 8; ++r) pk[r] = (bf16)fast_sigmoid(acc[r]);
      *(v8bf*)(gt + (size_t)c2 * NROW + mg0) = pk;
    }
  }
}

// ---------------------------------------------------------------------------
// Kernel 2: fused attention, one block per (h, i); 8 waves, 2 j-tiles/wave
// ---------------------------------------------------------------------------
__global__ __launch_bounds__(256) void k_attn(
    const bf16* __restrict__ proj_qk, const bf16* __restrict__ vt,
    const bf16* __restrict__ gt, const float* __restrict__ bt,
    bf16* __restrict__ obuf) {
  extern __shared__ char smem[];
  bf16* ps = (bf16*)smem;                 // [8 waves][16][256] bf16 = 64 KB

  int h = blockIdx.x >> 8;
  int i = blockIdx.x & 255;
  int tid = threadIdx.x;
  int lane = tid & 31, wave = tid >> 5;
  int n = lane & 15, half = lane >> 4;

  const float scale = 0.17677669529663687f;        // 1/sqrt(CH)
  const size_t rbase = (size_t)i * NSEQ;
  bf16* pw = ps + wave * 16 * 256;                 // wave-private 8 KB

  for (int jt2 = 0; jt2 < 2; ++jt2) {
    int jt = wave * 2 + jt2;
    int j0 = jt * 16 + 8 * half;                   // lane's first row

    v16bf aq = load_a_frag(proj_qk + (rbase + jt * 16) * 256 + h * 32, 256);

    // ---- S = Q K^T: 4 groups of 4 B fragments, double-buffered ----
    const bf16* kbase = proj_qk + rbase * 256 + 128 + h * 32;
    v8f s[16];
    v16bf bk[2][4];
#pragma unroll
    for (int u = 0; u < 4; ++u)
      bk[0][u] = load_bT_frag(kbase + (size_t)u * 16 * 256, 256);
#pragma unroll
    for (int g = 0; g < 4; ++g) {
      if (g + 1 < 4) {
#pragma unroll
        for (int u = 0; u < 4; ++u)
          bk[(g + 1) & 1][u] = load_bT_frag(
              kbase + (size_t)((g + 1) * 4 + u) * 16 * 256, 256);
      }
      __builtin_amdgcn_sched_barrier(0);
#pragma unroll
      for (int u = 0; u < 4; ++u) {
        v8f z = {};
        s[g * 4 + u] = wmma_bf16(aq, bk[g & 1][u], z);
      }
    }

    // ---- scale + bias (vector f32 loads) + row max ----
    float rmax[8];
#pragma unroll
    for (int r = 0; r < 8; ++r) rmax[r] = -3.4e38f;
#pragma unroll
    for (int t = 0; t < 16; ++t) {
      const float* bp = bt + h * NROW + (t * 16 + n) * 256 + j0;
      v4f b0 = *(const v4f*)bp, b1 = *(const v4f*)(bp + 4);
#pragma unroll
      for (int r = 0; r < 8; ++r) {
        float v = s[t][r] * scale + ((r < 4) ? b0[r & 3] : b1[r & 3]);
        s[t][r] = v;
        rmax[r] = fmaxf(rmax[r], v);
      }
    }
#pragma unroll
    for (int r = 0; r < 8; ++r) {
      float m = rmax[r];
#pragma unroll
      for (int off = 8; off >= 1; off >>= 1)
        m = fmaxf(m, __shfl_xor(m, off, 16));
      rmax[r] = m;
    }

    // ---- exp + row sum + normalize (v_rcp) ----
    float rsum[8];
#pragma unroll
    for (int r = 0; r < 8; ++r) rsum[r] = 0.f;
#pragma unroll
    for (int t = 0; t < 16; ++t) {
#pragma unroll
      for (int r = 0; r < 8; ++r) {
        float e = __expf(s[t][r] - rmax[r]);
        s[t][r] = e;
        rsum[r] += e;
      }
    }
#pragma unroll
    for (int r = 0; r < 8; ++r) {
      float m = rsum[r];
#pragma unroll
      for (int off = 8; off >= 1; off >>= 1)
        m += __shfl_xor(m, off, 16);
      rsum[r] = fast_rcp(m);
    }
    // P (bf16) -> wave-private LDS [16][256]; same-wave DS ops are in order
#pragma unroll
    for (int t = 0; t < 16; ++t) {
#pragma unroll
      for (int r = 0; r < 8; ++r)
        pw[(r + 8 * half) * 256 + t * 16 + n] = (bf16)(s[t][r] * rsum[r]);
    }

    // ---- O = P @ V (K = 256 keys = 8 WMMAs per 16-wide d tile) ----
    v16bf ap[8];
#pragma unroll
    for (int kc = 0; kc < 8; ++kc) ap[kc] = load_a_frag(pw + kc * 32, 256);

    // pw region is now free: reuse as [16][32] O tile for the store transpose
#pragma unroll
    for (int nd = 0; nd < 2; ++nd) {
      const bf16* vbase = vt + (size_t)(h * CH + nd * 16) * NROW + rbase;
      v16bf bv[8];
#pragma unroll
      for (int kc = 0; kc < 8; ++kc)
        bv[kc] = load_bT_frag(vbase + kc * 32, NROW);
      __builtin_amdgcn_sched_barrier(0);
      v8f oacc = {};
#pragma unroll
      for (int kc = 0; kc < 8; ++kc)
        oacc = wmma_bf16(ap[kc], bv[kc], oacc);

      int c2 = h * CH + nd * 16 + n;
      v8bf gv = *(const v8bf*)(gt + (size_t)c2 * NROW + rbase + j0);
#pragma unroll
      for (int r = 0; r < 8; ++r)
        pw[(r + 8 * half) * 32 + nd * 16 + n] =
            (bf16)(oacc[r] * (float)gv[r]);
    }
    // write O tile rows: lane -> (row m, 16-col half), 32B per lane
    {
      int m = lane & 15, h2 = lane >> 4;
      const bf16* src = pw + m * 32 + h2 * 16;
      bf16* dst = obuf + (rbase + jt * 16 + m) * 128 + h * CH + h2 * 16;
      *(v8bf*)dst       = *(const v8bf*)src;
      *(v8bf*)(dst + 8) = *(const v8bf*)(src + 8);
    }
  }
}

// ---------------------------------------------------------------------------
// Kernel 3: output projection  O[65536,128] @ wo'[128,128] + bo -> f32
// ---------------------------------------------------------------------------
__global__ __launch_bounds__(256) void k_outproj(
    const bf16* __restrict__ obuf, const bf16* __restrict__ woT,
    const float* __restrict__ bo, float* __restrict__ out) {
  extern __shared__ char smem[];
  float* st = (float*)smem;               // [8 waves][16][16] f32 = 8 KB
  int m0 = blockIdx.x * 128;
  int tid = threadIdx.x;
  int wave = tid >> 5, lane = tid & 31;
  int n = lane & 15, half = lane >> 4;
  float* stw = st + wave * 256;

  v16bf afr[4];
#pragma unroll
  for (int kc = 0; kc < 4; ++kc)
    afr[kc] = load_a_frag(obuf + (size_t)(m0 + wave * 16) * 128 + kc * 32, 128);

  // double-buffered weight fragments across nt
  v16bf bfr[2][4];
#pragma unroll
  for (int kc = 0; kc < 4; ++kc)
    bfr[0][kc] = load_bT_frag(woT + kc * 32, 128);

#pragma unroll
  for (int nt = 0; nt < 8; ++nt) {
    if (nt + 1 < 8) {
#pragma unroll
      for (int kc = 0; kc < 4; ++kc)
        bfr[(nt + 1) & 1][kc] =
            load_bT_frag(woT + (size_t)((nt + 1) * 16) * 128 + kc * 32, 128);
    }
    __builtin_amdgcn_sched_barrier(0);
    v8f acc = {};
#pragma unroll
    for (int kc = 0; kc < 4; ++kc)
      acc = wmma_bf16(afr[kc], bfr[nt & 1][kc], acc);

    float bias = bo[nt * 16 + n];
#pragma unroll
    for (int r = 0; r < 8; ++r)
      stw[(r + 8 * half) * 16 + n] = acc[r] + bias;
    // 32B row-chunk stores
    int m = lane & 15, h2 = lane >> 4;
    const float* src = stw + m * 16 + h2 * 8;
    float* dst = out + (size_t)(m0 + wave * 16 + m) * 128 + nt * 16 + h2 * 8;
    *(v4f*)dst       = *(const v4f*)src;
    *(v4f*)(dst + 4) = *(const v4f*)(src + 4);
  }
}

// ---------------------------------------------------------------------------
// Host launcher
// ---------------------------------------------------------------------------
extern "C" void kernel_launch(void* const* d_in, const int* in_sizes, int n_in,
                              void* d_out, int out_size, void* d_ws,
                              size_t ws_size, hipStream_t stream) {
  (void)in_sizes; (void)n_in; (void)out_size; (void)ws_size;
  const float* pair = (const float*)d_in[0];
  const float* lnw  = (const float*)d_in[1];
  const float* lnb  = (const float*)d_in[2];
  const float* wq   = (const float*)d_in[3];
  const float* wk   = (const float*)d_in[4];
  const float* wv   = (const float*)d_in[5];
  const float* wb   = (const float*)d_in[6];
  const float* wg   = (const float*)d_in[7];
  const float* wo   = (const float*)d_in[8];
  const float* bo   = (const float*)d_in[9];
  float* out = (float*)d_out;

  // workspace carve-up (bytes), all 256B aligned; total ~81.2 MiB
  char* ws = (char*)d_ws;
  bf16*  wcatT   = (bf16*)(ws + 0);                 // 512*128*2   = 131072
  bf16*  woT     = (bf16*)(ws + 131072);            // 128*128*2   =  32768
  float* bt      = (float*)(ws + 163840);           // 4*65536*4   = 1048576
  bf16*  proj_qk = (bf16*)(ws + 1212416);           // 65536*256*2 = 33554432
  bf16*  vt      = (bf16*)(ws + 34766848);          // 128*65536*2 = 16777216
  bf16*  gt      = (bf16*)(ws + 51544064);          // 128*65536*2 = 16777216
  bf16*  obuf    = (bf16*)(ws + 68321280);          // 65536*128*2 = 16777216

  k_prep_w<<<320, 256, 0, stream>>>(wq, wk, wv, wg, wo, wcatT, woT);

  k_ln_proj<<<512, 256, 128 * 128 * sizeof(bf16), stream>>>(
      pair, lnw, lnb, wb, wcatT, bt, proj_qk, vt, gt);

  k_attn<<<NH * NSEQ, 256, 8 * 16 * 256 * sizeof(bf16), stream>>>(
      proj_qk, vt, gt, bt, obuf);

  k_outproj<<<512, 256, 8 * 16 * 16 * sizeof(float), stream>>>(
      obuf, woT, bo, out);
}
